// VSformer_86973087744529
// MI455X (gfx1250) — matrix operand order repaired and checked by reading, not compiled, not run on hardware
//
#include <hip/hip_runtime.h>

// Problem constants (from reference): B,C,T,V = 32,256,64,48
#define BB 32
#define CC 256
#define TT 64
#define VV 48
#define GG 3
#define CC4 64
#define NTOK (BB*TT*VV)   // 98304 tokens; V=48=3x16 so 16-row tiles never cross t

typedef unsigned short u16;
typedef __attribute__((ext_vector_type(16))) __bf16 v16bf;
typedef __attribute__((ext_vector_type(8)))  float  v8f;

__device__ __forceinline__ u16 f2bf(float x){
  union { float f; unsigned u; } c; c.f = x;
  unsigned u = c.u + 0x7fffu + ((c.u >> 16) & 1u);   // round-to-nearest-even
  return (u16)(u >> 16);
}

__device__ __forceinline__ void tok2btv(int n, int& b, int& t, int& v){
  v = n % VV; t = (n / VV) % TT; b = n / (VV * TT);
}

// A fragment 16x32 bf16 from row-major [M x ldk].
// ISA layout: lanes 0-15 row M=lane, K = {0..7, 16..23}; lanes 16-31 same rows, K = {8..15, 24..31}.
// -> per lane: two contiguous 16-byte loads.
__device__ __forceinline__ v16bf load_fragA(const u16* __restrict__ A, int ldk, int row0, int k0){
  int lane = threadIdx.x & 31;
  int r  = row0 + (lane & 15);
  int kb = k0 + ((lane < 16) ? 0 : 8);
  const u16* p = A + (size_t)r * ldk + kb;
  union { uint4 u[2]; v16bf v; } f;
  f.u[0] = *reinterpret_cast<const uint4*>(p);
  f.u[1] = *reinterpret_cast<const uint4*>(p + 16);
  return f.v;
}

// B fragment 32x16 bf16 from N-major (pre-transposed) weights Bt[n*ldk + k].
// ISA layout: lanes 0-15 column N=lane, K = {0..7, 16..23}; lanes 16-31 same columns, K = {8..15, 24..31}.
__device__ __forceinline__ v16bf load_fragB(const u16* __restrict__ Bt, int ldk, int n0, int k0){
  int lane = threadIdx.x & 31;
  int n  = n0 + (lane & 15);
  int kb = k0 + ((lane < 16) ? 0 : 8);
  const u16* p = Bt + (size_t)n * ldk + kb;
  union { uint4 u[2]; v16bf v; } f;
  f.u[0] = *reinterpret_cast<const uint4*>(p);
  f.u[1] = *reinterpret_cast<const uint4*>(p + 16);
  return f.v;
}

#define WMMA_BF16(a, b, c) \
  __builtin_amdgcn_wmma_f32_16x16x32_bf16(false, (a), false, (b), (short)0, (c), false, false)

// Generic WMMA GEMM, register-blocked: one wave computes a 32x64 tile of
// D = A(bf16) x Bt(bf16) + bias  (2 M-subtiles x 4 N-subtiles, 8 accumulators).
// Per k-step: 2 A-fragment loads + 4 B-fragment loads feed 8 back-to-back WMMAs.
// MODE 0: f32 row-major out
// MODE 1: f32 row-major out + residual add from BCTV aux (proj + skip)
// MODE 2: bf16 row-major out
// MODE 3: bf16 row-major out through exact GELU
// MODE 4: f32 BCTV out (channel offset)
// MODE 5: f32 BCTV out + add from row-major aux (final residual)
template<int MODE>
__global__ void __launch_bounds__(32)
k_gemm(const u16* __restrict__ A, const u16* __restrict__ Bt,
       const float* __restrict__ bias, int Kd, int Nd,
       float* __restrict__ outf, u16* __restrict__ outb,
       const float* __restrict__ aux, int Ctot, int chOff)
{
  const int m0 = blockIdx.x * 32;
  const int n0 = blockIdx.y * 64;
  v8f acc[2][4] = {};
  for (int k0 = 0; k0 < Kd; k0 += 32){
    v16bf a0 = load_fragA(A, Kd, m0,      k0);
    v16bf a1 = load_fragA(A, Kd, m0 + 16, k0);
    #pragma unroll
    for (int j = 0; j < 4; ++j){
      v16bf b = load_fragB(Bt, Kd, n0 + 16 * j, k0);
      acc[0][j] = WMMA_BF16(a0, b, acc[0][j]);
      acc[1][j] = WMMA_BF16(a1, b, acc[1][j]);
    }
  }
  const int lane = threadIdx.x & 31;
  const int half = (lane < 16) ? 0 : 8;
  #pragma unroll
  for (int mi = 0; mi < 2; ++mi){
    #pragma unroll
    for (int j = 0; j < 4; ++j){
      const int col = n0 + 16 * j + (lane & 15);
      const float bv = bias ? bias[col] : 0.0f;
      #pragma unroll
      for (int r = 0; r < 8; ++r){
        int row = m0 + 16 * mi + r + half;  // D layout: vgpr r -> M=r (lanes 0-15) / r+8 (16-31)
        float v = acc[mi][j][r] + bv;
        if (MODE == 0){
          outf[(size_t)row * Nd + col] = v;
        } else if (MODE == 1){
          int b_, t_, v_; tok2btv(row, b_, t_, v_);
          v += aux[(((size_t)b_ * Ctot + col) * TT + t_) * VV + v_];
          outf[(size_t)row * Nd + col] = v;
        } else if (MODE == 2){
          outb[(size_t)row * Nd + col] = f2bf(v);
        } else if (MODE == 3){
          float g = 0.5f * v * (1.0f + erff(v * 0.70710678118654752f)); // exact GELU
          outb[(size_t)row * Nd + col] = f2bf(g);
        } else if (MODE == 4){
          int b_, t_, v_; tok2btv(row, b_, t_, v_);
          outf[(((size_t)b_ * Ctot + chOff + col) * TT + t_) * VV + v_] = v;
        } else {
          v += aux[(size_t)row * Nd + col];
          int b_, t_, v_; tok2btv(row, b_, t_, v_);
          outf[(((size_t)b_ * Ctot + chOff + col) * TT + t_) * VV + v_] = v;
        }
      }
    }
  }
}

// Causal temporal conv (k=5, pad (4,0)), 192->64 channels, as kt-shifted WMMA GEMMs.
// M-tile stays 16 (uniform t-shift per tile; 48 % 32 != 0), but all 4 N-subtiles are
// register-blocked so each A fragment feeds 4 back-to-back WMMAs.
__global__ void __launch_bounds__(32)
k_tclconv(const u16* __restrict__ Arow,   // [NTOK x 192] bf16 token-major
          const u16* __restrict__ Wp,     // [5][64][192] bf16, N-major per tap
          const float* __restrict__ bias, // [64]
          float* __restrict__ ycat, int chOff)
{
  const int m0 = blockIdx.x * 16;
  int b  = m0 / (TT * VV);
  int rr = m0 % (TT * VV);
  int t  = rr / VV;
  int v0 = rr % VV;
  v8f acc[4] = {};
  for (int kt = 0; kt < 5; ++kt){
    int ts = t + kt - 4;                 // causal: pad 4 on the left
    if (ts < 0 || ts >= TT) continue;
    int rowb = (b * TT + ts) * VV + v0;
    const u16* Wk = Wp + (size_t)kt * (CC4 * 192);
    for (int k0 = 0; k0 < 192; k0 += 32){
      v16bf a = load_fragA(Arow, 192, rowb, k0);
      #pragma unroll
      for (int j = 0; j < 4; ++j){
        v16bf w = load_fragB(Wk, 192, 16 * j, k0);
        acc[j] = WMMA_BF16(a, w, acc[j]);
      }
    }
  }
  const int lane = threadIdx.x & 31;
  const int half = (lane < 16) ? 0 : 8;
  #pragma unroll
  for (int j = 0; j < 4; ++j){
    const int co = 16 * j + (lane & 15);
    const float bv = bias[co];
    #pragma unroll
    for (int r = 0; r < 8; ++r){
      int vv = v0 + r + half;
      ycat[(((size_t)b * CC + chOff + co) * TT + t) * VV + vv] = acc[j][r] + bv;
    }
  }
}

// LayerNorm over C=256: one wave32 per token, 8 channels/lane, shuffle reduction. Output bf16 rows.
__global__ void __launch_bounds__(256)
k_ln(const float* __restrict__ src, const float* __restrict__ gam,
     const float* __restrict__ bet, u16* __restrict__ dst, int bctv)
{
  int wid  = threadIdx.x >> 5;
  int lane = threadIdx.x & 31;
  int n = blockIdx.x * 8 + wid;
  int b, t, v; tok2btv(n, b, t, v);
  float x[8]; float s = 0.f, ss = 0.f;
  #pragma unroll
  for (int i = 0; i < 8; ++i){
    int c = lane + 32 * i;
    float xv = bctv ? src[(((size_t)b * CC + c) * TT + t) * VV + v]
                    : src[(size_t)n * CC + c];
    x[i] = xv; s += xv; ss += xv * xv;
  }
  #pragma unroll
  for (int o = 16; o > 0; o >>= 1){
    s  += __shfl_xor(s,  o, 32);
    ss += __shfl_xor(ss, o, 32);
  }
  float mean = s * (1.0f / CC);
  float var  = ss * (1.0f / CC) - mean * mean;   // population variance (ddof=0)
  float rstd = rsqrtf(var + 1e-5f);
  #pragma unroll
  for (int i = 0; i < 8; ++i){
    int c = lane + 32 * i;
    dst[(size_t)n * CC + c] = f2bf((x[i] - mean) * rstd * gam[c] + bet[c]);
  }
}

// y0: grouped V-mixing einsum 'bgctu,gvu->bgctv' (48x48 per group of 8 channels). 0.6 GF -> VALU.
__global__ void k_y0(const float* __restrict__ f, const float* __restrict__ gconv,
                     float* __restrict__ ycat)
{
  int idx = blockIdx.x * 256 + threadIdx.x;
  if (idx >= BB * CC4 * TT * VV) return;
  int v = idx % VV;
  int t = (idx / VV) % TT;
  int c = (idx / (VV * TT)) % CC4;
  int b = idx / (VV * TT * CC4);
  int g = c >> 3;
  const float* fr = f + (((size_t)b * CC + c) * TT + t) * VV;
  const float* gm = gconv + ((size_t)g * VV + v) * VV;
  float s = 0.f;
  #pragma unroll 8
  for (int u = 0; u < VV; ++u) s += fr[u] * gm[u];
  ycat[(((size_t)b * CC + c) * TT + t) * VV + v] = s;
}

// y1: grouped temporal conv k=7, pad (3,3), 8 in-ch per group. 0.7 GF -> VALU.
__global__ void k_y1(const float* __restrict__ f, const float* __restrict__ w,
                     const float* __restrict__ bias, float* __restrict__ ycat)
{
  int idx = blockIdx.x * 256 + threadIdx.x;
  if (idx >= BB * CC4 * TT * VV) return;
  int v = idx % VV;
  int t = (idx / VV) % TT;
  int c = (idx / (VV * TT)) % CC4;
  int b = idx / (VV * TT * CC4);
  int g = c >> 3;
  float s = bias[c];
  for (int j = 0; j < 8; ++j){
    const float* fr = f + (((size_t)b * CC + CC4 + g * 8 + j) * TT) * VV;
    const float* wr = w + ((size_t)c * 8 + j) * 7;
    #pragma unroll
    for (int kt = 0; kt < 7; ++kt){
      int ts = t + kt - 3;
      if (ts < 0 || ts >= TT) continue;
      s += fr[ts * VV + v] * wr[kt];
    }
  }
  ycat[(((size_t)b * CC + CC4 + c) * TT + t) * VV + v] = s;
}

// Pack attention input rows: row n=(b,t,v) with v=g*16+u, 64 channels from f[ch0..ch0+63]. bf16.
__global__ void k_pack_attn(const float* __restrict__ f, u16* __restrict__ apack, int ch0)
{
  int idx = blockIdx.x * 256 + threadIdx.x;
  if (idx >= NTOK * CC4) return;
  int c = idx & 63;
  int n = idx >> 6;
  int b, t, v; tok2btv(n, b, t, v);
  apack[idx] = f2bf(f[(((size_t)b * CC + ch0 + c) * TT + t) * VV + v]);
}

// Attention core per (b,t,g): 4 heads x 16 positions x 16 dim; scale NH^-0.5 = 0.5. Output bf16 rows.
__global__ void __launch_bounds__(64)
k_attn_core(const float* __restrict__ qkv, u16* __restrict__ ao)
{
  __shared__ float tile[16][193];        // 16 positions x 192 (q|k|v), padded
  int bid = blockIdx.x;
  int g = bid % GG;
  int t = (bid / GG) % TT;
  int b = bid / (GG * TT);
  int base = (b * TT + t) * VV + g * 16;
  for (int i = threadIdx.x; i < 16 * 192; i += 64){
    int u = i / 192, c = i % 192;
    tile[u][c] = qkv[(size_t)(base + u) * 192 + c];
  }
  __syncthreads();
  int nh = threadIdx.x >> 4;
  int u  = threadIdx.x & 15;
  float q[16];
  #pragma unroll
  for (int d = 0; d < 16; ++d) q[d] = tile[u][nh * 16 + d];
  float sc[16]; float mx = -3.0e38f;
  #pragma unroll
  for (int w = 0; w < 16; ++w){
    float s = 0.f;
    #pragma unroll
    for (int d = 0; d < 16; ++d) s += q[d] * tile[w][64 + nh * 16 + d];
    s *= 0.5f;
    sc[w] = s; mx = fmaxf(mx, s);
  }
  float sum = 0.f;
  #pragma unroll
  for (int w = 0; w < 16; ++w){ sc[w] = expf(sc[w] - mx); sum += sc[w]; }
  float inv = 1.0f / sum;
  u16* dst = ao + (size_t)(base + u) * CC4 + nh * 16;
  #pragma unroll
  for (int d = 0; d < 16; ++d){
    float o = 0.f;
    #pragma unroll
    for (int w = 0; w < 16; ++w) o += sc[w] * tile[w][128 + nh * 16 + d];
    dst[d] = f2bf(o * inv);
  }
}

// Pack concatenated y (BCTV f32) into token-major bf16 rows for the proj GEMM.
__global__ void k_pack_y(const float* __restrict__ ycat, u16* __restrict__ yrow)
{
  int idx = blockIdx.x * 256 + threadIdx.x;
  if (idx >= NTOK * CC) return;
  int c = idx & 255;
  int n = idx >> 8;
  int b, t, v; tok2btv(n, b, t, v);
  yrow[idx] = f2bf(ycat[(((size_t)b * CC + c) * TT + t) * VV + v]);
}

// Weight pack: row-major [K x N] f32 -> N-major bf16 (so B fragments are K-contiguous).
__global__ void k_pack_wT(const float* __restrict__ w, u16* __restrict__ out, int K, int N)
{
  int idx = blockIdx.x * 256 + threadIdx.x;
  if (idx >= K * N) return;
  int n = idx / K, k = idx % K;
  out[idx] = f2bf(w[(size_t)k * N + n]);
}

// tcl weight pack: (64, 192, 5, 1) f32 -> [kt][co][ci] bf16.
__global__ void k_pack_tcl(const float* __restrict__ w, u16* __restrict__ out)
{
  int idx = blockIdx.x * 256 + threadIdx.x;
  if (idx >= 5 * CC4 * 192) return;
  int ci = idx % 192;
  int co = (idx / 192) % CC4;
  int kt = idx / (192 * CC4);
  out[((size_t)kt * CC4 + co) * 192 + ci] = f2bf(w[((size_t)co * 192 + ci) * 5 + kt]);
}

extern "C" void kernel_launch(void* const* d_in, const int* in_sizes, int n_in,
                              void* d_out, int out_size, void* d_ws, size_t ws_size,
                              hipStream_t stream)
{
  (void)in_sizes; (void)n_in; (void)out_size; (void)ws_size;
  const float* input   = (const float*)d_in[0];
  const float* n1g     = (const float*)d_in[1];
  const float* n1b     = (const float*)d_in[2];
  const float* map_w   = (const float*)d_in[3];
  const float* map_b   = (const float*)d_in[4];
  const float* gconv   = (const float*)d_in[5];
  const float* tconv_w = (const float*)d_in[6];
  const float* tconv_b = (const float*)d_in[7];
  const float* qkv_w0  = (const float*)d_in[8];
  const float* qkv_b0  = (const float*)d_in[9];
  const float* apr_w0  = (const float*)d_in[10];
  const float* apr_b0  = (const float*)d_in[11];
  const float* tcl_w0  = (const float*)d_in[12];
  const float* tcl_b0  = (const float*)d_in[13];
  const float* qkv_w1  = (const float*)d_in[14];
  const float* qkv_b1  = (const float*)d_in[15];
  const float* apr_w1  = (const float*)d_in[16];
  const float* apr_b1  = (const float*)d_in[17];
  const float* tcl_w1  = (const float*)d_in[18];
  const float* tcl_b1  = (const float*)d_in[19];
  const float* proj_w  = (const float*)d_in[20];
  const float* proj_b  = (const float*)d_in[21];
  const float* n2g     = (const float*)d_in[22];
  const float* n2b     = (const float*)d_in[23];
  const float* mlp_w1  = (const float*)d_in[24];
  const float* mlp_b1  = (const float*)d_in[25];
  const float* mlp_w2  = (const float*)d_in[26];
  const float* mlp_b2  = (const float*)d_in[27];
  float* outp = (float*)d_out;
  char* ws = (char*)d_ws;

  // Workspace layout with lifetime-based aliasing (~442 MB peak).
  const size_t szN = (size_t)NTOK;
  size_t o_xn    = 0;                         // [N x 256] bf16   (later reused as hin)
  size_t o_f     = o_xn    + szN * CC  * 2;   // [B,256,T,V] f32  (later reused as out_buf)
  size_t o_ycat  = o_f     + szN * CC  * 4;   // [B,256,T,V] f32  -- start of region reused as h
  size_t o_qkv   = o_ycat  + szN * CC  * 4;   // [N x 192] f32    -- part of h region
  size_t o_apack = o_qkv   + szN * 192 * 4;   // [N x 64] bf16    -- part of h region
  size_t o_ao    = o_apack + szN * 64  * 2;   // [N x 64] bf16    -- end of h region (exactly N*1024*2 total)
  size_t o_yrow  = o_ao    + szN * 64  * 2;   // [N x 256] bf16
  size_t o_a0row = o_yrow  + szN * CC  * 2;   // [N x 192] bf16
  size_t o_w     = o_a0row + szN * 192 * 2;   // packed bf16 weights (~1.7 MB)

  u16*   xn     = (u16*)  (ws + o_xn);
  float* f      = (float*)(ws + o_f);
  float* ycat   = (float*)(ws + o_ycat);
  float* qkv    = (float*)(ws + o_qkv);
  u16*   apack  = (u16*)  (ws + o_apack);
  u16*   ao     = (u16*)  (ws + o_ao);
  u16*   yrow   = (u16*)  (ws + o_yrow);
  u16*   a0row  = (u16*)  (ws + o_a0row);
  float* outbuf = (float*)(ws + o_f);     // alias: f dead after attn packs / y0 / y1
  u16*   hin    = (u16*)  (ws + o_xn);    // alias: xn dead after map GEMM
  u16*   hbuf   = (u16*)  (ws + o_ycat);  // alias: ycat/qkv/apack/ao dead after proj GEMM

  u16* mapT  = (u16*)(ws + o_w);
  u16* qkvT0 = mapT  + 65536;
  u16* aprT0 = qkvT0 + 12288;
  u16* qkvT1 = aprT0 + 12288;
  u16* aprT1 = qkvT1 + 12288;
  u16* tclP0 = aprT1 + 12288;
  u16* tclP1 = tclP0 + 61440;
  u16* projT = tclP1 + 61440;
  u16* mlp1T = projT + 65536;
  u16* mlp2T = mlp1T + 262144;

  dim3 b32(32), b64(64), b256(256);
  auto cdiv = [](int a, int b){ return (a + b - 1) / b; };

  // ---- weight precision packs (bf16, N-major) ----
  k_pack_wT<<<cdiv(65536,256),  b256, 0, stream>>>(map_w,  mapT,  256, 256);
  k_pack_wT<<<cdiv(12288,256),  b256, 0, stream>>>(qkv_w0, qkvT0, 64, 192);
  k_pack_wT<<<cdiv(12288,256),  b256, 0, stream>>>(apr_w0, aprT0, 64, 192);
  k_pack_wT<<<cdiv(12288,256),  b256, 0, stream>>>(qkv_w1, qkvT1, 64, 192);
  k_pack_wT<<<cdiv(12288,256),  b256, 0, stream>>>(apr_w1, aprT1, 64, 192);
  k_pack_tcl<<<cdiv(61440,256), b256, 0, stream>>>(tcl_w0, tclP0);
  k_pack_tcl<<<cdiv(61440,256), b256, 0, stream>>>(tcl_w1, tclP1);
  k_pack_wT<<<cdiv(65536,256),  b256, 0, stream>>>(proj_w, projT, 256, 256);
  k_pack_wT<<<cdiv(262144,256), b256, 0, stream>>>(mlp_w1, mlp1T, 256, 1024);
  k_pack_wT<<<cdiv(262144,256), b256, 0, stream>>>(mlp_w2, mlp2T, 1024, 256);

  // ---- LN1 + map GEMM (N x 256 x 256) -> f in (B,C,T,V) ----
  k_ln<<<NTOK/8, b256, 0, stream>>>(input, n1g, n1b, xn, 1);
  k_gemm<4><<<dim3(NTOK/32, 4), b32, 0, stream>>>(xn, mapT, map_b, 256, 256, f, nullptr, nullptr, CC, 0);

  // ---- y0 (V-mix) and y1 (grouped k7 conv) ----
  int ne = BB * CC4 * TT * VV;
  k_y0<<<cdiv(ne,256), b256, 0, stream>>>(f, gconv, ycat);
  k_y1<<<cdiv(ne,256), b256, 0, stream>>>(f, tconv_w, tconv_b, ycat);

  // ---- attention branch 0: pack -> qkv GEMM -> softmax core -> out-proj GEMM -> causal conv GEMM ----
  k_pack_attn<<<cdiv(NTOK*64,256), b256, 0, stream>>>(f, apack, 2*CC4);
  k_gemm<0><<<dim3(NTOK/32, 3), b32, 0, stream>>>(apack, qkvT0, qkv_b0, 64, 192, qkv, nullptr, nullptr, 0, 0);
  k_attn_core<<<BB*TT*GG, b64, 0, stream>>>(qkv, ao);
  k_gemm<2><<<dim3(NTOK/32, 3), b32, 0, stream>>>(ao, aprT0, apr_b0, 64, 192, nullptr, a0row, nullptr, 0, 0);
  k_tclconv<<<NTOK/16, b32, 0, stream>>>(a0row, tclP0, tcl_b0, ycat, 2*CC4);

  // ---- attention branch 1 ----
  k_pack_attn<<<cdiv(NTOK*64,256), b256, 0, stream>>>(f, apack, 3*CC4);
  k_gemm<0><<<dim3(NTOK/32, 3), b32, 0, stream>>>(apack, qkvT1, qkv_b1, 64, 192, qkv, nullptr, nullptr, 0, 0);
  k_attn_core<<<BB*TT*GG, b64, 0, stream>>>(qkv, ao);
  k_gemm<2><<<dim3(NTOK/32, 3), b32, 0, stream>>>(ao, aprT1, apr_b1, 64, 192, nullptr, a0row, nullptr, 0, 0);
  k_tclconv<<<NTOK/16, b32, 0, stream>>>(a0row, tclP1, tcl_b1, ycat, 3*CC4);

  // ---- proj GEMM + residual (skip read straight from input in BCTV) ----
  k_pack_y<<<cdiv(NTOK*CC,256), b256, 0, stream>>>(ycat, yrow);
  k_gemm<1><<<dim3(NTOK/32, 4), b32, 0, stream>>>(yrow, projT, proj_b, 256, 256, outbuf, nullptr, input, CC, 0);

  // ---- MLP: LN2 -> GEMM+GELU (N x 256 x 1024) -> GEMM + residual -> d_out in (B,C,T,V) ----
  k_ln<<<NTOK/8, b256, 0, stream>>>(outbuf, n2g, n2b, hin, 0);
  k_gemm<3><<<dim3(NTOK/32, 16), b32, 0, stream>>>(hin, mlp1T, mlp_b1, 256, 1024, nullptr, hbuf, nullptr, 0, 0);
  k_gemm<5><<<dim3(NTOK/32, 4), b32, 0, stream>>>(hbuf, mlp2T, mlp_b2, 1024, 256, outp, nullptr, outbuf, CC, 0);
}
